// RNN_64776696758363
// MI455X (gfx1250) — compile-verified, matrix operand order checked by reading, not run on hardware
//
#include <hip/hip_runtime.h>

// ---------------------------------------------------------------- types
typedef __bf16  bf16_t;
typedef __attribute__((ext_vector_type(16))) __bf16 v16bf;
typedef __attribute__((ext_vector_type(8)))  float  v8f;

// ---------------------------------------------------------------- sizes
constexpr int T    = 512;
constexpr int B    = 64;
constexpr int DIN  = 256;
constexpr int DLAT = 512;

// workspace layout (bytes)
constexpr size_t XIN_OFF  = 0;                                   // T*B*DLAT fp32
constexpr size_t XIN_BYT  = (size_t)T * B * DLAT * 4;            // 64 MB
constexpr size_t WHBF_OFF = XIN_OFF + XIN_BYT;                   // DLAT*DLAT bf16
constexpr size_t WHBF_BYT = (size_t)DLAT * DLAT * 2;             // 512 KB
constexpr size_t CTR_OFF  = WHBF_OFF + WHBF_BYT;                 // 4 counters
constexpr size_t XB_OFF   = CTR_OFF + 256;                       // x in bf16
constexpr size_t XB_BYT   = (size_t)T * B * DIN * 2;             // 16 MB
constexpr size_t WIB_OFF  = XB_OFF + XB_BYT;                     // W_in in bf16
constexpr size_t WIB_BYT  = (size_t)DLAT * DIN * 2;              // 256 KB
constexpr size_t HB_OFF   = WIB_OFF + WIB_BYT;                   // h bf16, double buffered
constexpr size_t HB_BYT   = (size_t)2 * B * DLAT * 2;            // 128 KB

// recurrence tiling: 4 batch groups x 4 column WGs
constexpr int NB_GROUPS = 4;     // 16 batch rows each
constexpr int NC_WGS    = 4;     // 128 output columns each
constexpr int COLS_PER_WG = DLAT / NC_WGS;                       // 128
constexpr int SMEM_WH   = COLS_PER_WG * DLAT * 2;                // 128 KB
constexpr int SMEM_H    = 16 * DLAT * 2;                         // 16 KB
constexpr int SMEM_BYTES = SMEM_WH + SMEM_H;                     // 147456 (<320KB/WGP)

// ---------------------------------------------------------------- helpers
__device__ __forceinline__ unsigned pack2_bf16(float lo, float hi) {
    unsigned short a = __builtin_bit_cast(unsigned short, (__bf16)lo);
    unsigned short b = __builtin_bit_cast(unsigned short, (__bf16)hi);
    return (unsigned)a | ((unsigned)b << 16);
}

// CDNA5 async copy: global -> LDS, 16 bytes per lane, tracked by ASYNCcnt.
// GVS mode: 64-bit SGPR base + per-lane 32-bit byte offset; vdst = LDS byte addr.
__device__ __forceinline__ void async_ld_b128(unsigned lds_addr, const void* sbase,
                                              unsigned goff) {
    asm volatile("global_load_async_to_lds_b128 %0, %1, %2 scope:SCOPE_DEV"
                 :: "v"(lds_addr), "v"(goff), "s"(sbase) : "memory");
}
__device__ __forceinline__ void wait_async0() {
    asm volatile("s_wait_asynccnt 0" ::: "memory");
}
__device__ __forceinline__ unsigned lds_off(const void* p) {
    return (unsigned)(size_t)p;          // generic->LDS: low 32 bits = LDS offset
}

// ---------------------------------------------------------------- kernel 0: zero barrier counters
__global__ void rnn_init_kernel(unsigned* ctr) {
    if (threadIdx.x < NB_GROUPS) ctr[threadIdx.x] = 0u;
}

// ---------------------------------------------------------------- kernel 1: fp32 -> bf16 (packed, 4 elems/thread)
__global__ void rnn_cvt_kernel(const float* __restrict__ src, unsigned* __restrict__ dst) {
    int i = blockIdx.x * blockDim.x + threadIdx.x;     // one float4 per thread
    float4 f = ((const float4*)src)[i];
    uint2 o;
    o.x = pack2_bf16(f.x, f.y);
    o.y = pack2_bf16(f.z, f.w);
    ((uint2*)dst)[i] = o;
}

// ---------------------------------------------------------------- kernel 2: xin = x @ W_in^T + b_h
// M = T*B = 32768, N = DLAT = 512, K = DIN = 256, operands pre-converted bf16.
__global__ __launch_bounds__(256) void rnn_xin_gemm_kernel(
    const bf16_t* __restrict__ xb,    // [T*B, DIN] bf16
    const bf16_t* __restrict__ wib,   // [DLAT, DIN] bf16
    const float*  __restrict__ b_h,   // [DLAT]
    float* __restrict__ xin)          // [T*B, DLAT]
{
    const int wave  = threadIdx.x >> 5;
    const int lane  = threadIdx.x & 31;
    const int l15   = lane & 15;
    const int khalf = (lane >> 4) & 1;         // 0: K 0..15, 1: K 16..31
    const int m0    = blockIdx.x * 16;
    const int n0    = wave * 64;

    v8f acc[4] = {};
#pragma unroll
    for (int k0 = 0; k0 < DIN; k0 += 32) {
        v16bf a = *(const v16bf*)(xb + (size_t)(m0 + l15) * DIN + k0 + khalf * 16);
#pragma unroll
        for (int nn = 0; nn < 4; ++nn) {
            v16bf b = *(const v16bf*)(wib + (size_t)(n0 + nn * 16 + l15) * DIN + k0 + khalf * 16);
            acc[nn] = __builtin_amdgcn_wmma_f32_16x16x32_bf16(
                false, a, false, b, (short)0, acc[nn], false, false);
        }
    }
#pragma unroll
    for (int nn = 0; nn < 4; ++nn) {
        const int col  = n0 + nn * 16 + l15;
        const float bi = b_h[col];
#pragma unroll
        for (int r = 0; r < 8; ++r) {
            const int row = m0 + r + khalf * 8;
            xin[(size_t)row * DLAT + col] = acc[nn][r] + bi;
        }
    }
}

// ---------------------------------------------------------------- kernel 3: recurrence
// 16 persistent WGs = 4 batch groups x 4 column WGs. W_h slice (128x512 bf16)
// LDS-resident. h carried as bf16 double buffer in global; each step the 16x512
// bf16 h tile is ASYNC-copied global->LDS (no VGPR/VALU), WMMA'd against the
// resident W_h slice, relu'd with xin, written to d_out (fp32) + hb (bf16).
__global__ __launch_bounds__(256) void rnn_recur_kernel(
    const float*  __restrict__ xin,   // [T*B, DLAT] (bias folded in)
    const bf16_t* __restrict__ whb,   // [DLAT, DLAT] bf16 row-major (j,k)
    bf16_t*       __restrict__ hb,    // [2][B, DLAT] bf16 h chain
    unsigned*     __restrict__ ctr,   // [NB_GROUPS]
    float*        __restrict__ out)   // [T, B, DLAT]
{
    extern __shared__ __align__(16) char smem[];
    bf16_t* wh_s = (bf16_t*)smem;                  // [128][512] bf16 (B operand slices)
    bf16_t* h_s  = (bf16_t*)(smem + SMEM_WH);      // [16][512]  bf16 (A operand)

    const int group = blockIdx.x / NC_WGS;         // batch group -> rows b0..b0+15
    const int cwg   = blockIdx.x % NC_WGS;         // column group -> cols j0..j0+127
    const int b0    = group * 16;
    const int j0    = cwg * COLS_PER_WG;

    const int tid   = threadIdx.x;
    const int wave  = tid >> 5;
    const int lane  = tid & 31;
    const int l15   = lane & 15;
    const int khalf = (lane >> 4) & 1;

    const unsigned wh_s_base = lds_off(wh_s);
    const unsigned h_s_base  = lds_off(h_s);

    // ---- async-load resident W_h slice (128KB contiguous) ----
#pragma unroll 4
    for (int i = tid; i < SMEM_WH / 16; i += 256)
        async_ld_b128(wh_s_base + i * 16, whb, (unsigned)j0 * DLAT * 2 + i * 16);
    wait_async0();
    __syncthreads();

    unsigned* my_ctr = ctr + group;

    for (int t = 0; t < T; ++t) {
        // ---- async-load h_t tile: rows b0..b0+15, bf16, 16KB contiguous ----
        const unsigned hb_byt = (unsigned)((t & 1) * B + b0) * DLAT * 2;
#pragma unroll
        for (int i = tid; i < SMEM_H / 16; i += 256)
            async_ld_b128(h_s_base + i * 16, hb, hb_byt + i * 16);
        wait_async0();
        __syncthreads();

        // ---- 16x16 tile per wave, K = 512 ----
        const int nloc = wave * 16;                // local column base within slice
        v8f acc = {};
#pragma unroll
        for (int k0 = 0; k0 < DLAT; k0 += 32) {
            v16bf a = *(const v16bf*)(h_s  + (size_t)l15 * DLAT + k0 + khalf * 16);
            v16bf b = *(const v16bf*)(wh_s + (size_t)(nloc + l15) * DLAT + k0 + khalf * 16);
            acc = __builtin_amdgcn_wmma_f32_16x16x32_bf16(
                false, a, false, b, (short)0, acc, false, false);
        }

        // ---- h_{t+1} = relu(acc + xin) -> d_out[t] fp32 + hb[(t+1)&1] bf16 ----
        const int col = j0 + nloc + l15;
        bf16_t* hb_next = hb + (size_t)((t + 1) & 1) * B * DLAT;
#pragma unroll
        for (int r = 0; r < 8; ++r) {
            const int brow = b0 + r + khalf * 8;
            const size_t idx = ((size_t)t * B + brow) * DLAT + col;
            const float v = fmaxf(acc[r] + xin[idx], 0.0f);
            out[idx] = v;
            hb_next[(size_t)brow * DLAT + col] = (bf16_t)v;
        }

        // ---- inter-WG barrier within batch group ----
        __threadfence();               // flush this wave's stores to device scope
        __syncthreads();
        if (tid == 0) {
            atomicAdd(my_ctr, 1u);
            const unsigned tgt = (unsigned)NC_WGS * (unsigned)(t + 1);
            while (__hip_atomic_load(my_ctr, __ATOMIC_ACQUIRE,
                                     __HIP_MEMORY_SCOPE_AGENT) < tgt) {
                __builtin_amdgcn_s_sleep(2);
            }
        }
        __syncthreads();
    }
}

// ---------------------------------------------------------------- launcher
extern "C" void kernel_launch(void* const* d_in, const int* in_sizes, int n_in,
                              void* d_out, int out_size, void* d_ws, size_t ws_size,
                              hipStream_t stream) {
    const float* x    = (const float*)d_in[0];   // [T,B,DIN]
    const float* h0   = (const float*)d_in[1];   // [B,DLAT]
    const float* w_in = (const float*)d_in[2];   // [DLAT,DIN]
    const float* w_h  = (const float*)d_in[3];   // [DLAT,DLAT]
    const float* b_h  = (const float*)d_in[4];   // [DLAT]
    float*       out  = (float*)d_out;

    float*    xin = (float*)((char*)d_ws + XIN_OFF);
    bf16_t*   whb = (bf16_t*)((char*)d_ws + WHBF_OFF);
    unsigned* ctr = (unsigned*)((char*)d_ws + CTR_OFF);
    bf16_t*   xb  = (bf16_t*)((char*)d_ws + XB_OFF);
    bf16_t*   wib = (bf16_t*)((char*)d_ws + WIB_OFF);
    bf16_t*   hb  = (bf16_t*)((char*)d_ws + HB_OFF);

    static_assert(SMEM_BYTES <= 320 * 1024, "LDS budget");
    hipFuncSetAttribute((const void*)rnn_recur_kernel,
                        hipFuncAttributeMaxDynamicSharedMemorySize, SMEM_BYTES);

    rnn_init_kernel<<<1, 32, 0, stream>>>(ctr);
    // one-pass fp32 -> bf16 conversions (4 elems/thread)
    rnn_cvt_kernel<<<(T * B * DIN) / 4 / 256, 256, 0, stream>>>(x, (unsigned*)xb);
    rnn_cvt_kernel<<<(DLAT * DIN) / 4 / 256, 256, 0, stream>>>(w_in, (unsigned*)wib);
    rnn_cvt_kernel<<<(DLAT * DLAT) / 4 / 256, 256, 0, stream>>>(w_h, (unsigned*)whb);
    rnn_cvt_kernel<<<(B * DLAT) / 4 / 256, 256, 0, stream>>>(h0, (unsigned*)hb); // hb[0]=bf16(h0)

    rnn_xin_gemm_kernel<<<(T * B) / 16, 256, 0, stream>>>(xb, wib, b_h, xin);
    rnn_recur_kernel<<<NB_GROUPS * NC_WGS, 256, SMEM_BYTES, stream>>>(
        xin, whb, hb, ctr, out);
}
